// FlashSelfAttention_21414706938602
// MI455X (gfx1250) — matrix operand order repaired and checked by reading, not compiled, not run on hardware
//
#include <hip/hip_runtime.h>
#include <hip/hip_bf16.h>
#include <math.h>

// ---------------------------------------------------------------------------
// FlashSelfAttention for MI455X (gfx1250, wave32, WMMA bf16)
// B=4, T=2048, C=1024, H=16, D=64.  fp32 in/out, bf16 WMMA internally.
// ---------------------------------------------------------------------------

#define BB 4
#define TT 2048
#define CC 1024
#define HH 16
#define DD 64

typedef __attribute__((ext_vector_type(16))) __bf16 v16bf;
typedef __attribute__((ext_vector_type(8)))  __bf16 v8bf;
typedef __attribute__((ext_vector_type(4)))  __bf16 v4bf;
typedef __attribute__((ext_vector_type(8)))  float  v8f;

static __device__ __forceinline__ v8f vzero8f() {
    v8f z = {0.f, 0.f, 0.f, 0.f, 0.f, 0.f, 0.f, 0.f};
    return z;
}

// Load a 16-element bf16 fragment as two 16B (b128) loads at byte-aligned
// element offsets o1 / o2 from p.
static __device__ __forceinline__ v16bf ld_frag(const __bf16* p, int o1, int o2) {
    v8bf lo = *reinterpret_cast<const v8bf*>(p + o1);
    v8bf hi = *reinterpret_cast<const v8bf*>(p + o2);
    return __builtin_shufflevector(lo, hi, 0, 1, 2, 3, 4, 5, 6, 7,
                                           8, 9, 10, 11, 12, 13, 14, 15);
}

static __device__ __forceinline__ v8f wmma_bf16(v16bf a, v16bf b, v8f c) {
    // D = A(16x32) * B(32x16) + C, f32 accumulate
    return __builtin_amdgcn_wmma_f32_16x16x32_bf16(false, a, false, b,
                                                   (short)0, c, false, false);
}

// ---------------------------------------------------------------------------
// Kernel: fp32 -> bf16 elementwise (x conversion), 4 elems / thread
// ---------------------------------------------------------------------------
__global__ __launch_bounds__(256) void cvt_f32_bf16(const float* __restrict__ in,
                                                    __bf16* __restrict__ out,
                                                    int n) {
    int i = (blockIdx.x * 256 + threadIdx.x) * 4;
    if (i + 3 < n) {
        float4 v = *reinterpret_cast<const float4*>(in + i);
        v4bf o = { (__bf16)v.x, (__bf16)v.y, (__bf16)v.z, (__bf16)v.w };
        *reinterpret_cast<v4bf*>(out + i) = o;
    }
}

// ---------------------------------------------------------------------------
// Kernel: W [K][N] f32  ->  Wt [N][K] bf16  (LDS 32x32 tile transpose)
// ---------------------------------------------------------------------------
__global__ __launch_bounds__(256) void transpose_w(const float* __restrict__ in,
                                                   __bf16* __restrict__ out,
                                                   int K, int N) {
    __shared__ float tile[32][33];
    const int n0 = blockIdx.x * 32;
    const int k0 = blockIdx.y * 32;
    const int tx = threadIdx.x;   // 0..31
    const int ty = threadIdx.y;   // 0..7
#pragma unroll
    for (int i = 0; i < 32; i += 8)
        tile[ty + i][tx] = in[(size_t)(k0 + ty + i) * N + n0 + tx];
    __syncthreads();
#pragma unroll
    for (int i = 0; i < 32; i += 8)
        out[(size_t)(n0 + ty + i) * K + k0 + tx] = (__bf16)tile[tx][ty + i];
}

// ---------------------------------------------------------------------------
// Kernel: QKV GEMM.  M=8192 (b*T+t), N=3072, K=1024.
//   A = xb [M][K] bf16 row-major, B = Wa_t [N][K] bf16 ("N-major").
//   Epilogue: + bias, scatter (branch hoisted: `which` is block-uniform):
//     which==0 (q-proj) -> Aq [B,H,T,D]   (plays KEY   role in attention)
//     which==1 (k-proj) -> Ak [B,H,T,D]   (plays QUERY role in attention)
//     which==2 (v-proj) -> Vt [B,H,D,T]   (transposed for P@V B-fragments)
// Each wave: 32x64 tile = 2x4 WMMA accumulators. Block = 8 waves = 128x128.
// ---------------------------------------------------------------------------
__global__ __launch_bounds__(256) void qkv_gemm(const __bf16* __restrict__ Xb,
                                                const __bf16* __restrict__ Wt,
                                                const float*  __restrict__ bias,
                                                __bf16* __restrict__ Aq,
                                                __bf16* __restrict__ Ak,
                                                __bf16* __restrict__ Vt) {
    const int lane = threadIdx.x & 31;
    const int wv   = threadIdx.x >> 5;
    const int lc   = lane & 15;
    const int hiq  = lane >> 4;
    const int m0 = blockIdx.x * 128 + (wv >> 1) * 32;
    const int n0 = blockIdx.y * 128 + (wv & 1) * 64;
    const int kb = hiq * 16;   // A-frag k base
    const int ob = hiq * 16;   // B-frag k base

    v8f acc[2][4];
#pragma unroll
    for (int mi = 0; mi < 2; ++mi)
#pragma unroll
        for (int ni = 0; ni < 4; ++ni) acc[mi][ni] = vzero8f();

    for (int k0 = 0; k0 < CC; k0 += 32) {
        v16bf a[2], bf[4];
#pragma unroll
        for (int mi = 0; mi < 2; ++mi)
            a[mi] = ld_frag(Xb + (size_t)(m0 + mi * 16 + lc) * CC + k0, kb, kb + 8);
#pragma unroll
        for (int ni = 0; ni < 4; ++ni)
            bf[ni] = ld_frag(Wt + (size_t)(n0 + ni * 16 + lc) * CC + k0, ob, ob + 8);
#pragma unroll
        for (int mi = 0; mi < 2; ++mi)
#pragma unroll
            for (int ni = 0; ni < 4; ++ni)
                acc[mi][ni] = wmma_bf16(a[mi], bf[ni], acc[mi][ni]);
    }

    // ---- epilogue: bias add + bf16 convert into a register tile ----
    __bf16 res[2][4][8];
#pragma unroll
    for (int ni = 0; ni < 4; ++ni) {
        const float bv = bias[n0 + ni * 16 + lc];
#pragma unroll
        for (int mi = 0; mi < 2; ++mi)
#pragma unroll
            for (int j = 0; j < 8; ++j)
                res[mi][ni][j] = (__bf16)(acc[mi][ni][j] + bv);
    }

    // ---- scatter; `which` is uniform for the whole block ----
    const int which = n0 >> 10;       // 0=q-proj, 1=k-proj, 2=v-proj
    const int c0    = n0 & 1023;      // col within [0,C)
    const int h     = c0 >> 6;        // wave's 64-col span = exactly one head
    const int b     = m0 >> 11;       // 128-row tile lies in one batch (T=2048)
    const int t0    = m0 & 2047;
    const size_t bh = (size_t)(b * HH + h);

    if (which == 2) {
        // Vt [B,H,D,T]: per element, row d = ni*16+lc, col t
        __bf16* dst = Vt + (bh * DD) * TT + t0;
#pragma unroll
        for (int ni = 0; ni < 4; ++ni) {
            const int d = ni * 16 + lc;
#pragma unroll
            for (int mi = 0; mi < 2; ++mi)
#pragma unroll
                for (int j = 0; j < 8; ++j)
                    dst[(size_t)d * TT + mi * 16 + j + hiq * 8] = res[mi][ni][j];
        }
    } else {
        __bf16* dst = (which == 0 ? Aq : Ak) + (bh * TT + t0) * DD;
#pragma unroll
        for (int mi = 0; mi < 2; ++mi)
#pragma unroll
            for (int j = 0; j < 8; ++j) {
                const size_t rowoff = (size_t)(mi * 16 + j + hiq * 8) * DD;
#pragma unroll
                for (int ni = 0; ni < 4; ++ni)
                    dst[rowoff + ni * 16 + lc] = res[mi][ni][j];
            }
    }
}

// ---------------------------------------------------------------------------
// Kernel: causal flash attention, one wave per 16 query rows of one (b,h).
//   Q = k-projection, K = q-projection (reference swaps them), V transposed.
//   Per 32-key step: 4 WMMA for S = Q*K^T, online softmax (shfl row stats),
//   P repacked C-layout -> A-layout via per-wave LDS, 4 WMMA for O += P*V.
// ---------------------------------------------------------------------------
__global__ __launch_bounds__(256) void flash_attn(const __bf16* __restrict__ Qp,
                                                  const __bf16* __restrict__ Kp,
                                                  const __bf16* __restrict__ Vt,
                                                  __bf16* __restrict__ Yb) {
    __shared__ __align__(16) __bf16 plds[8 * 640];   // 16 rows x 40 bf16 / wave
    const int lane = threadIdx.x & 31;
    const int wv   = threadIdx.x >> 5;
    const int lc   = lane & 15;
    const int hiq  = lane >> 4;
    const int gw   = blockIdx.x * 8 + wv;   // 8192 waves total
    const int bh   = gw >> 7;               // (b*H + h)
    const int m0   = (gw & 127) << 4;       // query tile start in [0,T)
    const int b    = bh >> 4;
    const int h    = bh & 15;

    const __bf16* Q = Qp + (size_t)bh * TT * DD;
    const __bf16* K = Kp + (size_t)bh * TT * DD;
    const __bf16* V = Vt + (size_t)bh * DD * TT;
    __bf16* lp = plds + wv * 640;

    const int kb = hiq * 16;   // A-frag k base
    const int ob = hiq * 16;   // B-frag k base

    // Q A-fragments for d = [0,32) and [32,64)
    v16bf qf[2];
#pragma unroll
    for (int f = 0; f < 2; ++f)
        qf[f] = ld_frag(Q + (size_t)(m0 + lc) * DD + f * 32, kb, kb + 8);

    v8f o[4];
#pragma unroll
    for (int dc = 0; dc < 4; ++dc) o[dc] = vzero8f();
    float mrow[8], lrow[8];
#pragma unroll
    for (int j = 0; j < 8; ++j) { mrow[j] = -1e30f; lrow[j] = 0.f; }

    const float LOG2E = 1.44269504f;
    for (int j0 = 0; j0 < m0 + 16; j0 += 32) {
        // ---- S = (Q * K^T) * 1/sqrt(D), two 16-key tiles ----
        v8f s[2];
#pragma unroll
        for (int t = 0; t < 2; ++t) {
            const __bf16* kp = K + (size_t)(j0 + t * 16 + lc) * DD;
            v16bf kf0 = ld_frag(kp, ob, ob + 8);        // d 0..31
            v16bf kf1 = ld_frag(kp + 32, ob, ob + 8);   // d 32..63
            v8f c = vzero8f();
            c = wmma_bf16(qf[0], kf0, c);
            c = wmma_bf16(qf[1], kf1, c);
            s[t] = c;
        }
        // ---- scale + causal mask ----
#pragma unroll
        for (int j = 0; j < 8; ++j) {
            const int row = m0 + j + hiq * 8;
            float v0 = s[0][j] * 0.125f;
            float v1 = s[1][j] * 0.125f;
            if (j0 + lc > row)      v0 = -1e30f;
            if (j0 + 16 + lc > row) v1 = -1e30f;
            s[0][j] = v0;
            s[1][j] = v1;
        }
        // ---- online softmax: row stats across 16 lanes of each half ----
        float alpha[8];
#pragma unroll
        for (int j = 0; j < 8; ++j) {
            float rm = fmaxf(s[0][j], s[1][j]);
#pragma unroll
            for (int off = 8; off >= 1; off >>= 1)
                rm = fmaxf(rm, __shfl_xor(rm, off, 16));
            const float nm = fmaxf(mrow[j], rm);
            alpha[j] = exp2f((mrow[j] - nm) * LOG2E);
            const float p0 = exp2f((s[0][j] - nm) * LOG2E);
            const float p1 = exp2f((s[1][j] - nm) * LOG2E);
            float rs = p0 + p1;
#pragma unroll
            for (int off = 8; off >= 1; off >>= 1)
                rs += __shfl_xor(rs, off, 16);
            lrow[j] = lrow[j] * alpha[j] + rs;
            mrow[j] = nm;
            // stage P (bf16) into LDS in C-layout coordinates
            const int r = j + hiq * 8;
            lp[r * 40 + lc]      = (__bf16)p0;
            lp[r * 40 + 16 + lc] = (__bf16)p1;
        }
        // same-wave DS ordering barrier before A-layout reload
        asm volatile("s_wait_dscnt 0x0" ::: "memory");
        v16bf pf = ld_frag(lp + lc * 40, kb, kb + 8);
        // ---- O = O*alpha + P @ V ----
#pragma unroll
        for (int dc = 0; dc < 4; ++dc) {
#pragma unroll
            for (int j = 0; j < 8; ++j) o[dc][j] *= alpha[j];
            const __bf16* vp = V + (size_t)(dc * 16 + lc) * TT + j0;
            v16bf vf = ld_frag(vp, ob, ob + 8);
            o[dc] = wmma_bf16(pf, vf, o[dc]);
        }
    }

    // ---- normalize + store Y [B*T][C] bf16 ----
#pragma unroll
    for (int j = 0; j < 8; ++j) {
        const float inv = 1.0f / lrow[j];
        const int row = m0 + j + hiq * 8;
        const size_t base = ((size_t)b * TT + row) * CC + h * DD;
#pragma unroll
        for (int dc = 0; dc < 4; ++dc)
            Yb[base + dc * 16 + lc] = (__bf16)(o[dc][j] * inv);
    }
}

// ---------------------------------------------------------------------------
// Kernel: output projection GEMM.  M=8192, N=1024, K=1024, f32 out + bias.
// ---------------------------------------------------------------------------
__global__ __launch_bounds__(256) void proj_gemm(const __bf16* __restrict__ Yb,
                                                 const __bf16* __restrict__ Wt,
                                                 const float*  __restrict__ bias,
                                                 float* __restrict__ out) {
    const int lane = threadIdx.x & 31;
    const int wv   = threadIdx.x >> 5;
    const int lc   = lane & 15;
    const int hiq  = lane >> 4;
    const int m0 = blockIdx.x * 128 + (wv >> 1) * 32;
    const int n0 = blockIdx.y * 128 + (wv & 1) * 64;
    const int kb = hiq * 16;
    const int ob = hiq * 16;

    v8f acc[2][4];
#pragma unroll
    for (int mi = 0; mi < 2; ++mi)
#pragma unroll
        for (int ni = 0; ni < 4; ++ni) acc[mi][ni] = vzero8f();

    for (int k0 = 0; k0 < CC; k0 += 32) {
        v16bf a[2], bf[4];
#pragma unroll
        for (int mi = 0; mi < 2; ++mi)
            a[mi] = ld_frag(Yb + (size_t)(m0 + mi * 16 + lc) * CC + k0, kb, kb + 8);
#pragma unroll
        for (int ni = 0; ni < 4; ++ni)
            bf[ni] = ld_frag(Wt + (size_t)(n0 + ni * 16 + lc) * CC + k0, ob, ob + 8);
#pragma unroll
        for (int mi = 0; mi < 2; ++mi)
#pragma unroll
            for (int ni = 0; ni < 4; ++ni)
                acc[mi][ni] = wmma_bf16(a[mi], bf[ni], acc[mi][ni]);
    }

#pragma unroll
    for (int ni = 0; ni < 4; ++ni) {
        const int n = n0 + ni * 16 + lc;
        const float bv = bias[n];
#pragma unroll
        for (int mi = 0; mi < 2; ++mi) {
#pragma unroll
            for (int j = 0; j < 8; ++j) {
                const int m = m0 + mi * 16 + j + hiq * 8;
                out[(size_t)m * CC + n] = acc[mi][ni][j] + bv;
            }
        }
    }
}

// ---------------------------------------------------------------------------
// Host launcher
// ---------------------------------------------------------------------------
extern "C" void kernel_launch(void* const* d_in, const int* in_sizes, int n_in,
                              void* d_out, int out_size, void* d_ws, size_t ws_size,
                              hipStream_t stream) {
    (void)in_sizes; (void)n_in; (void)out_size; (void)ws_size;
    const float* x      = (const float*)d_in[0];  // [B,T,C]
    const float* W_attn = (const float*)d_in[1];  // [C,3C]
    const float* b_attn = (const float*)d_in[2];  // [3C]
    const float* W_proj = (const float*)d_in[3];  // [C,C]
    const float* b_proj = (const float*)d_in[4];  // [C]
    float* out = (float*)d_out;                   // [B,T,C] f32

    // workspace carve-out (256B aligned slices); total ~88 MB
    char* ws = (char*)d_ws;
    auto carve = [&](size_t bytes) {
        char* p = ws;
        ws += (bytes + 255) & ~(size_t)255;
        return p;
    };
    const size_t M = (size_t)BB * TT;  // 8192
    __bf16* xb  = (__bf16*)carve(M * CC * 2);              // x in bf16
    __bf16* Wat = (__bf16*)carve((size_t)3 * CC * CC * 2); // W_attn^T [3C][C]
    __bf16* Wpt = (__bf16*)carve((size_t)CC * CC * 2);     // W_proj^T [C][C]
    __bf16* Aq  = (__bf16*)carve(M * CC * 2);              // q-proj [B,H,T,D]
    __bf16* Ak  = (__bf16*)carve(M * CC * 2);              // k-proj [B,H,T,D]
    __bf16* Vt  = (__bf16*)carve(M * CC * 2);              // v-proj [B,H,D,T]
    __bf16* Yb  = (__bf16*)carve(M * CC * 2);              // attn out [M][C]

    // 1) precision conversion / weight transposes
    {
        const int n = (int)(M * CC);
        cvt_f32_bf16<<<n / (256 * 4), 256, 0, stream>>>(x, xb, n);
        transpose_w<<<dim3(3 * CC / 32, CC / 32), dim3(32, 8), 0, stream>>>(
            W_attn, Wat, CC, 3 * CC);
        transpose_w<<<dim3(CC / 32, CC / 32), dim3(32, 8), 0, stream>>>(
            W_proj, Wpt, CC, CC);
    }
    // 2) fused QKV projection (+bias, head scatter)
    qkv_gemm<<<dim3(M / 128, 3 * CC / 128), 256, 0, stream>>>(
        xb, Wat, b_attn, Aq, Ak, Vt);
    // 3) causal flash attention; reference uses sdpa(k, q, v): query=k-proj
    flash_attn<<<dim3((BB * HH * (TT / 16)) / 8), 256, 0, stream>>>(
        /*Q=*/Ak, /*K=*/Aq, Vt, Yb);
    // 4) output projection (+bias), f32 result
    proj_gemm<<<dim3(M / 128, CC / 128), 256, 0, stream>>>(
        Yb, Wpt, b_proj, out);
}
